// CosineTransformerBlock_42932493091365
// MI455X (gfx1250) — matrix operand order repaired and checked by reading, not compiled, not run on hardware
//
#include <hip/hip_runtime.h>
#include <hip/hip_bf16.h>
#include <math.h>

// ---------------------------------------------------------------------------
// CosineTransformerBlock on MI455X (gfx1250).
//  - bf16 WMMA (v_wmma_f32_16x16x32_bf16) for all GEMMs
//  - softmax-free attention reassociated: (qn knT) v == qn (knT v)
//  - GEMM tiles staged via Tensor Data Mover (tensor_load_to_lds) with the
//    TDM pad feature producing the bank-conflict-free LDS stride, double
//    buffered and pipelined against WMMA via s_wait_tensorcnt.
// ---------------------------------------------------------------------------

typedef __attribute__((ext_vector_type(16))) __bf16 v16bf;
typedef __attribute__((ext_vector_type(8)))  __bf16 v8bf;
typedef __attribute__((ext_vector_type(8)))  float  v8f;
typedef __attribute__((ext_vector_type(4)))  unsigned int u32x4;
typedef __attribute__((ext_vector_type(8)))  int i32x8;
typedef __attribute__((ext_vector_type(4)))  int i32x4;

union VAB { v16bf v; v8bf h[2]; };

#define DIMC    1024
#define SEQ     2048
#define BATCH   4
#define NHEADS  16
#define DHEAD   64
#define MLPC    4096
#define ROWS    (BATCH*SEQ)   /* 8192 */
#define LDSW    40            /* LDS row stride in bf16 (80B): 64B row + 16B TDM pad */

__device__ __forceinline__ float wave_sum32(float x) {
#pragma unroll
  for (int off = 16; off; off >>= 1) x += __shfl_xor(x, off, 32);
  return x;
}

__device__ __forceinline__ v8f wmma_bf16(const VAB& a, const VAB& b, v8f c) {
  return __builtin_amdgcn_wmma_f32_16x16x32_bf16(false, a.v, false, b.v,
                                                 (short)0, c, false, false);
}

// ---- Tensor Data Mover: load a 2D tile (tile_d1 lines of tile_d0 bf16) from
// global (line stride = stride0 elements) into LDS, padding 16B after every
// 64B so the LDS row stride is 80B (= LDSW bf16 elements).
__device__ __forceinline__ void tdm_load_2d(const void* lds_ptr, const void* gptr,
                                            unsigned tensor_d0, unsigned tensor_d1,
                                            unsigned long long stride0,
                                            unsigned tile_d0, unsigned tile_d1) {
  unsigned lds = (unsigned)(uintptr_t)lds_ptr;          // aperture: low 32b = LDS offset
  unsigned long long ga = (unsigned long long)(uintptr_t)gptr;
  u32x4 g0;
  g0[0] = 1u;                                           // count=1, user descriptor
  g0[1] = lds;                                          // lds_addr
  g0[2] = (unsigned)ga;                                 // global_addr[31:0]
  g0[3] = (unsigned)((ga >> 32) & 0x01ffffffull) | (2u << 30); // addr[56:32] | type=2
  i32x8 g1;
  g1[0] = (int)((1u << 16)      /* data_size = 2B            */
              | (1u << 20)      /* pad_enable                */
              | (3u << 22)      /* pad_interval = 16 DWORDs  */
              | (3u << 25));    /* pad_amount   = 4 DWORDs   */
  g1[1] = (int)((tensor_d0 & 0xffffu) << 16);           // abar=0 | tensor_dim0 lo16
  g1[2] = (int)((tensor_d0 >> 16) | ((tensor_d1 & 0xffffu) << 16));
  g1[3] = (int)((tensor_d1 >> 16) | (tile_d0 << 16));   // tensor_dim1 hi | tile_dim0
  g1[4] = (int)(tile_d1 & 0xffffu);                     // tile_dim1 ; tile_dim2=0
  g1[5] = (int)(unsigned)(stride0 & 0xffffffffull);     // tensor_dim0_stride lo32
  g1[6] = (int)(unsigned)((stride0 >> 32) & 0xffffull); // stride hi16 | dim1_stride lo = 0
  g1[7] = 0;
  i32x4 g2; g2[0] = 1; g2[1] = 1; g2[2] = 0; g2[3] = 0; // tensor_dim2=1, tensor_dim3=1
  i32x4 g3; g3[0] = 0; g3[1] = (int)(1u << 16); g3[2] = 0; g3[3] = 0; // tensor_dim4=1
#if __clang_major__ >= 23
  i32x8 g4 = {0, 0, 0, 0, 0, 0, 0, 0};
  __builtin_amdgcn_tensor_load_to_lds(g0, g1, g2, g3, g4, 0);
#else
  __builtin_amdgcn_tensor_load_to_lds(g0, g1, g2, g3, 0);
#endif
}

// --------------------------- weight convert + transpose ---------------------
// w: (K x N) f32 row-major  ->  wt: (N x K) bf16 row-major
__global__ void __launch_bounds__(256)
cvtT_kernel(const float* __restrict__ w, __bf16* __restrict__ wt, int K, int N) {
  int i = blockIdx.x * 256 + threadIdx.x;
  if (i >= K * N) return;
  int n = i / K, k = i % K;
  wt[i] = (__bf16)w[(size_t)k * N + n];
}

// --------------------------- layernorm (f32 in, bf16 out) -------------------
__global__ void __launch_bounds__(256)
ln_kernel(const float* __restrict__ x, const float* __restrict__ g,
          const float* __restrict__ bb, __bf16* __restrict__ y, int rows) {
  int wave = threadIdx.x >> 5, lane = threadIdx.x & 31;
  int row = blockIdx.x * 8 + wave;
  if (row >= rows) return;
  const float* xr = x + (size_t)row * DIMC;
  float v[32];
  float s1 = 0.f, s2 = 0.f;
#pragma unroll
  for (int j = 0; j < 32; ++j) {
    float t = xr[j * 32 + lane];
    v[j] = t; s1 += t; s2 += t * t;
  }
  s1 = wave_sum32(s1); s2 = wave_sum32(s2);
  float mu  = s1 * (1.f / DIMC);
  float var = s2 * (1.f / DIMC) - mu * mu;
  float inv = rsqrtf(var + 1e-5f);
  __bf16* yr = y + (size_t)row * DIMC;
#pragma unroll
  for (int j = 0; j < 32; ++j) {
    int c = j * 32 + lane;
    yr[c] = (__bf16)((v[j] - mu) * inv * g[c] + bb[c]);
  }
}

// --------------------------- per-head L2 normalize (in place) ---------------
__global__ void __launch_bounds__(256)
l2norm64_kernel(__bf16* __restrict__ p, int groups) {
  int wave = threadIdx.x >> 5, lane = threadIdx.x & 31;
  int grp = blockIdx.x * 8 + wave;
  if (grp >= groups) return;
  __bf16* q = p + (size_t)grp * 64;
  float a = (float)q[lane * 2], b = (float)q[lane * 2 + 1];
  float ss = wave_sum32(a * a + b * b);
  float sc = 1.f / fmaxf(sqrtf(ss), 1e-12f);
  q[lane * 2]     = (__bf16)(a * sc);
  q[lane * 2 + 1] = (__bf16)(b * sc);
}

// --------------------------- main WMMA GEMM (TDM staged) --------------------
// C(MxN) = A(MxK,bf16) * Bt(NxK,bf16)^T ; 128x128 block, 8 waves of 32x64.
// Double-buffered LDS; wave 0 drives the Tensor Data Mover one tile ahead.
// EPI 0: out bf16 ; EPI 1: outF = res + C + bias (f32) ; EPI 2: bf16 gelu(C+bias)
template <int EPI>
__global__ void __launch_bounds__(256)
gemm_kernel(const __bf16* __restrict__ A, const __bf16* __restrict__ Bt,
            int K, int N,
            __bf16* __restrict__ outB, float* __restrict__ outF,
            const float* __restrict__ bias, const float* __restrict__ res) {
  __shared__ __bf16 As[2 * 128 * LDSW];
  __shared__ __bf16 Bs[2 * 128 * LDSW];
  const int tid = threadIdx.x, lane = tid & 31, wave = tid >> 5;
  const int m0 = blockIdx.y * 128, n0 = blockIdx.x * 128;
  const int mo = (wave & 3) * 32;   // wave grid 4 (M) x 2 (N)
  const int no = (wave >> 2) * 64;
  const __bf16* Agt = A  + (size_t)m0 * K;   // tile origin (row m0, col 0)
  const __bf16* Bgt = Bt + (size_t)n0 * K;

  v8f acc[2][4] = {};
  const int s8  = (lane >> 4) * 8;
  const int s16 = (lane >> 4) * 16;
  const int mr  = lane & 15;
  const int nk  = K >> 5;

  if (wave == 0) {                       // prologue: DMA tile 0 into buffer 0
    tdm_load_2d(&As[0], Agt,      K, 128, (unsigned long long)K, 32, 128);
    tdm_load_2d(&Bs[0], Bgt,      K, 128, (unsigned long long)K, 32, 128);
  }

  for (int j = 0; j < nk; ++j) {
    __syncthreads();                     // everyone done reading buf[(j+1)&1]
    if (wave == 0) {
      if (j + 1 < nk) {                  // DMA tile j+1 while tile j computes
        int nb = (j + 1) & 1;
        tdm_load_2d(&As[nb * 128 * LDSW], Agt + (j + 1) * 32,
                    K, 128, (unsigned long long)K, 32, 128);
        tdm_load_2d(&Bs[nb * 128 * LDSW], Bgt + (j + 1) * 32,
                    K, 128, (unsigned long long)K, 32, 128);
        __builtin_amdgcn_s_wait_tensorcnt((short)2);  // tile j landed (in-order)
      } else {
        __builtin_amdgcn_s_wait_tensorcnt((short)0);
      }
    }
    __syncthreads();                     // tile j visible to all waves

    const __bf16* Ab = &As[(j & 1) * 128 * LDSW];
    const __bf16* Bb = &Bs[(j & 1) * 128 * LDSW];
    VAB a[2], b[4];
#pragma unroll
    for (int mt = 0; mt < 2; ++mt) {
      int r = mo + mt * 16 + mr;
      a[mt].h[0] = *(const v8bf*)&Ab[r * LDSW + s8];
      a[mt].h[1] = *(const v8bf*)&Ab[r * LDSW + 16 + s8];
    }
#pragma unroll
    for (int nt = 0; nt < 4; ++nt) {
      int r = no + nt * 16 + mr;
      b[nt].h[0] = *(const v8bf*)&Bb[r * LDSW + s8];
      b[nt].h[1] = *(const v8bf*)&Bb[r * LDSW + 16 + s8];
    }
#pragma unroll
    for (int mt = 0; mt < 2; ++mt)
#pragma unroll
      for (int nt = 0; nt < 4; ++nt)
        acc[mt][nt] = wmma_bf16(a[mt], b[nt], acc[mt][nt]);
  }

  const int mr8 = (lane >> 4) * 8;
  const int nc  = lane & 15;
#pragma unroll
  for (int mt = 0; mt < 2; ++mt)
#pragma unroll
    for (int nt = 0; nt < 4; ++nt)
#pragma unroll
      for (int r = 0; r < 8; ++r) {
        int M  = m0 + mo + mt * 16 + mr8 + r;
        int Nc = n0 + no + nt * 16 + nc;
        size_t idx = (size_t)M * N + Nc;
        float c = acc[mt][nt][r];
        if (EPI == 0) {
          outB[idx] = (__bf16)c;
        } else if (EPI == 1) {
          outF[idx] = res[idx] + c + bias[Nc];
        } else {  // exact gelu
          float t = c + bias[Nc];
          outB[idx] = (__bf16)(0.5f * t * (1.f + erff(t * 0.70710678118654752f)));
        }
      }
}

// --------------------------- S = knT * v per (b,h), stored transposed -------
__global__ void __launch_bounds__(128)
kv_s_kernel(const __bf16* __restrict__ Kp, const __bf16* __restrict__ Vp,
            __bf16* __restrict__ St) {
  const int bh = blockIdx.x, b = bh >> 4, h = bh & 15;
  const __bf16* kb = Kp + (size_t)b * SEQ * DIMC + h * DHEAD;
  const __bf16* vb = Vp + (size_t)b * SEQ * DIMC + h * DHEAD;
  __shared__ __bf16 Kt[64 * LDSW];   // Kt[d][kk] = kn[kk][d]
  __shared__ __bf16 Vt[64 * LDSW];
  const int tid = threadIdx.x, lane = tid & 31, wave = tid >> 5;
  const int mo = (wave & 1) * 32, no = (wave >> 1) * 32;
  const int srow = tid >> 2;          // 0..31  (seq within tile)
  const int d0   = (tid & 3) * 16;    // head-dim chunk
  const int s8 = (lane >> 4) * 8, s16 = (lane >> 4) * 16, mr = lane & 15;
  v8f acc[2][2] = {};

  for (int kk0 = 0; kk0 < SEQ; kk0 += 32) {
    __syncthreads();
    const __bf16* kr = kb + (size_t)(kk0 + srow) * DIMC + d0;
    const __bf16* vr = vb + (size_t)(kk0 + srow) * DIMC + d0;
    v8bf k0v = *(const v8bf*)kr, k1v = *(const v8bf*)(kr + 8);
    v8bf v0v = *(const v8bf*)vr, v1v = *(const v8bf*)(vr + 8);
#pragma unroll
    for (int j = 0; j < 8; ++j) {
      Kt[(d0 + j)     * LDSW + srow] = k0v[j];
      Kt[(d0 + 8 + j) * LDSW + srow] = k1v[j];
      Vt[(d0 + j)     * LDSW + srow] = v0v[j];
      Vt[(d0 + 8 + j) * LDSW + srow] = v1v[j];
    }
    __syncthreads();

    VAB a[2], bf[2];
#pragma unroll
    for (int mt = 0; mt < 2; ++mt) {
      int r = mo + mt * 16 + mr;
      a[mt].h[0] = *(const v8bf*)&Kt[r * LDSW + s8];
      a[mt].h[1] = *(const v8bf*)&Kt[r * LDSW + 16 + s8];
    }
#pragma unroll
    for (int nt = 0; nt < 2; ++nt) {
      int r = no + nt * 16 + mr;
      bf[nt].h[0] = *(const v8bf*)&Vt[r * LDSW + s8];
      bf[nt].h[1] = *(const v8bf*)&Vt[r * LDSW + 16 + s8];
    }
#pragma unroll
    for (int mt = 0; mt < 2; ++mt)
#pragma unroll
      for (int nt = 0; nt < 2; ++nt)
        acc[mt][nt] = wmma_bf16(a[mt], bf[nt], acc[mt][nt]);
  }

  __bf16* so = St + (size_t)bh * (DHEAD * DHEAD);
  const int mr8 = (lane >> 4) * 8, nc = lane & 15;
#pragma unroll
  for (int mt = 0; mt < 2; ++mt)
#pragma unroll
    for (int nt = 0; nt < 2; ++nt)
#pragma unroll
      for (int r = 0; r < 8; ++r) {
        int M = mo + mt * 16 + mr8 + r;
        int Nc = no + nt * 16 + nc;
        so[Nc * DHEAD + M] = (__bf16)acc[mt][nt][r];  // store S^T
      }
}

// --------------------------- out = qn * S per (b,h) -------------------------
__global__ void __launch_bounds__(128)
attn_out_kernel(const __bf16* __restrict__ Qp, const __bf16* __restrict__ St,
                __bf16* __restrict__ AO) {
  const int bh = blockIdx.y, b = bh >> 4, h = bh & 15;
  const int m0 = blockIdx.x * 64;
  const int tid = threadIdx.x, lane = tid & 31, wave = tid >> 5;
  const __bf16* qb = Qp + (size_t)b * SEQ * DIMC + h * DHEAD;
  const __bf16* sb = St + (size_t)bh * (DHEAD * DHEAD);
  const int mr = lane & 15, s8 = (lane >> 4) * 8, s16 = (lane >> 4) * 16;
  const int M = m0 + wave * 16 + mr;
  v8f acc[4] = {};

#pragma unroll
  for (int kb = 0; kb < DHEAD; kb += 32) {
    VAB a, bf[4];
    a.h[0] = *(const v8bf*)(qb + (size_t)M * DIMC + kb + s8);
    a.h[1] = *(const v8bf*)(qb + (size_t)M * DIMC + kb + 16 + s8);
#pragma unroll
    for (int nt = 0; nt < 4; ++nt) {
      int Nc = nt * 16 + mr;
      bf[nt].h[0] = *(const v8bf*)(sb + Nc * DHEAD + kb + s8);
      bf[nt].h[1] = *(const v8bf*)(sb + Nc * DHEAD + kb + 16 + s8);
    }
#pragma unroll
    for (int nt = 0; nt < 4; ++nt) acc[nt] = wmma_bf16(a, bf[nt], acc[nt]);
  }

  __bf16* ao = AO + (size_t)b * SEQ * DIMC + h * DHEAD;
  const int mr8 = (lane >> 4) * 8, nc = lane & 15;
#pragma unroll
  for (int nt = 0; nt < 4; ++nt)
#pragma unroll
    for (int r = 0; r < 8; ++r) {
      int Mr = m0 + wave * 16 + mr8 + r;
      ao[(size_t)Mr * DIMC + nt * 16 + nc] = (__bf16)acc[nt][r];
    }
}

// ---------------------------------------------------------------------------
extern "C" void kernel_launch(void* const* d_in, const int* in_sizes, int n_in,
                              void* d_out, int out_size, void* d_ws, size_t ws_size,
                              hipStream_t stream) {
  (void)in_sizes; (void)n_in; (void)out_size; (void)ws_size;
  const float* Q   = (const float*)d_in[0];
  const float* Kin = (const float*)d_in[1];
  const float* Vin = (const float*)d_in[2];
  const float* wq  = (const float*)d_in[3];
  const float* wk  = (const float*)d_in[4];
  const float* wv  = (const float*)d_in[5];
  const float* wo  = (const float*)d_in[6];
  const float* bo  = (const float*)d_in[7];
  const float* w1  = (const float*)d_in[8];
  const float* b1  = (const float*)d_in[9];
  const float* w2  = (const float*)d_in[10];
  const float* b2  = (const float*)d_in[11];
  const float* g1  = (const float*)d_in[12];
  const float* be1 = (const float*)d_in[13];
  const float* g2  = (const float*)d_in[14];
  const float* be2 = (const float*)d_in[15];
  float* out = (float*)d_out;

  char* ws = (char*)d_ws;
  size_t off = 0;
  auto take = [&](size_t bytes) -> char* {
    char* p = ws + off;
    off += (bytes + 255) & ~(size_t)255;
    return p;
  };

  __bf16* WqT = (__bf16*)take((size_t)DIMC * DIMC * 2);
  __bf16* WkT = (__bf16*)take((size_t)DIMC * DIMC * 2);
  __bf16* WvT = (__bf16*)take((size_t)DIMC * DIMC * 2);
  __bf16* WoT = (__bf16*)take((size_t)DIMC * DIMC * 2);
  __bf16* W1T = (__bf16*)take((size_t)MLPC * DIMC * 2);
  __bf16* W2T = (__bf16*)take((size_t)DIMC * MLPC * 2);
  __bf16* Qn  = (__bf16*)take((size_t)ROWS * DIMC * 2);
  __bf16* Kn  = (__bf16*)take((size_t)ROWS * DIMC * 2);
  __bf16* Vn  = (__bf16*)take((size_t)ROWS * DIMC * 2);
  __bf16* Pq  = (__bf16*)take((size_t)ROWS * DIMC * 2);
  __bf16* Pk  = (__bf16*)take((size_t)ROWS * DIMC * 2);
  __bf16* Pv  = (__bf16*)take((size_t)ROWS * DIMC * 2);
  __bf16* St  = (__bf16*)take((size_t)BATCH * NHEADS * DHEAD * DHEAD * 2);
  float*  X   = (float*) take((size_t)ROWS * DIMC * 4);
  __bf16* H   = (__bf16*)take((size_t)ROWS * MLPC * 2);
  __bf16* AO  = Qn;   // Qn dead after q-projection
  __bf16* XN  = Kn;   // Kn dead after k-projection

  // 1) weight convert+transpose to bf16
  cvtT_kernel<<<(DIMC * DIMC + 255) / 256, 256, 0, stream>>>(wq, WqT, DIMC, DIMC);
  cvtT_kernel<<<(DIMC * DIMC + 255) / 256, 256, 0, stream>>>(wk, WkT, DIMC, DIMC);
  cvtT_kernel<<<(DIMC * DIMC + 255) / 256, 256, 0, stream>>>(wv, WvT, DIMC, DIMC);
  cvtT_kernel<<<(DIMC * DIMC + 255) / 256, 256, 0, stream>>>(wo, WoT, DIMC, DIMC);
  cvtT_kernel<<<(DIMC * MLPC + 255) / 256, 256, 0, stream>>>(w1, W1T, DIMC, MLPC);
  cvtT_kernel<<<(DIMC * MLPC + 255) / 256, 256, 0, stream>>>(w2, W2T, MLPC, DIMC);

  // 2) LN1 of Q, K, V -> bf16
  ln_kernel<<<ROWS / 8, 256, 0, stream>>>(Q,   g1, be1, Qn, ROWS);
  ln_kernel<<<ROWS / 8, 256, 0, stream>>>(Kin, g1, be1, Kn, ROWS);
  ln_kernel<<<ROWS / 8, 256, 0, stream>>>(Vin, g1, be1, Vn, ROWS);

  // 3) q/k/v projections (WMMA GEMM, TDM-staged)
  dim3 gP(DIMC / 128, ROWS / 128);
  gemm_kernel<0><<<gP, 256, 0, stream>>>(Qn, WqT, DIMC, DIMC, Pq, nullptr, nullptr, nullptr);
  gemm_kernel<0><<<gP, 256, 0, stream>>>(Kn, WkT, DIMC, DIMC, Pk, nullptr, nullptr, nullptr);
  gemm_kernel<0><<<gP, 256, 0, stream>>>(Vn, WvT, DIMC, DIMC, Pv, nullptr, nullptr, nullptr);

  // 4) l2-normalize q and k per head
  l2norm64_kernel<<<(ROWS * NHEADS) / 8, 256, 0, stream>>>(Pq, ROWS * NHEADS);
  l2norm64_kernel<<<(ROWS * NHEADS) / 8, 256, 0, stream>>>(Pk, ROWS * NHEADS);

  // 5) S = knT v  (softmax-free attention reassociation), then out = qn S
  kv_s_kernel<<<BATCH * NHEADS, 128, 0, stream>>>(Pk, Pv, St);
  attn_out_kernel<<<dim3(SEQ / 64, BATCH * NHEADS), 128, 0, stream>>>(Pq, St, AO);

  // 6) x = Q + out @ wo + bo
  gemm_kernel<1><<<gP, 256, 0, stream>>>(AO, WoT, DIMC, DIMC, nullptr, X, bo, Q);

  // 7) LN2 -> bf16
  ln_kernel<<<ROWS / 8, 256, 0, stream>>>(X, g2, be2, XN, ROWS);

  // 8) h = gelu(xn @ w1 + b1)
  gemm_kernel<2><<<dim3(MLPC / 128, ROWS / 128), 256, 0, stream>>>(XN, W1T, DIMC, MLPC, H, nullptr, b1, nullptr);

  // 9) y = x + h @ w2 + b2 -> d_out
  gemm_kernel<1><<<dim3(DIMC / 128, ROWS / 128), 256, 0, stream>>>(H, W2T, MLPC, DIMC, nullptr, out, b2, X);
}
